// Attention_28613072126828
// MI455X (gfx1250) — matrix operand order repaired and checked by reading, not compiled, not run on hardware
//
#include <hip/hip_runtime.h>
#include <hip/hip_bf16.h>

typedef __attribute__((ext_vector_type(16))) _Float16 v16h;
typedef __attribute__((ext_vector_type(8)))  _Float16 v8h;
typedef __attribute__((ext_vector_type(8)))  float    v8f;

#define SEQ 2048
#define DHEAD 1024
#define BM 32
#define BN 32

static __device__ __forceinline__ v8f wmma_f16(v16h a, v16h b, v8f c) {
    // v_wmma_f32_16x16x32_f16  (8 args: neg_a, A, neg_b, B, c_mod, C, reuse_a, reuse_b)
    return __builtin_amdgcn_wmma_f32_16x16x32_f16(false, a, false, b, (short)0, c, false, false);
}

// A-fragment (16x32 f16, MxK): lanes 0-15 -> M=lane, K {kb8..kb8+7, kb8+16..kb8+23}, kb8 = (lane>=16)*8
static __device__ __forceinline__ v16h load_frag_a(const _Float16* base, int row0,
                                                   int rowStrideH, int kOff, int lane) {
    const int m   = lane & 15;
    const int kb8 = (lane >> 4) << 3;
    const _Float16* p = base + (size_t)(row0 + m) * rowStrideH + kOff + kb8;
    v8h lo = *(const v8h*)p;
    v8h hi = *(const v8h*)(p + 16);
    return __builtin_shufflevector(lo, hi, 0,1,2,3,4,5,6,7,8,9,10,11,12,13,14,15);
}

// B-fragment (32x16 f16, KxN): lane -> N=lane&15, K {kb8..kb8+7, kb8+16..kb8+23}, kb8 = (lane>=16)*8 ("K-major" LDS)
static __device__ __forceinline__ v16h load_frag_b(const _Float16* base, int col0,
                                                   int rowStrideH, int kOff, int lane) {
    const int n    = lane & 15;
    const int kb16 = (lane >> 4) << 3;
    const _Float16* p = base + (size_t)(col0 + n) * rowStrideH + kOff + kb16;
    v8h lo = *(const v8h*)p;
    v8h hi = *(const v8h*)(p + 16);
    return __builtin_shufflevector(lo, hi, 0,1,2,3,4,5,6,7,8,9,10,11,12,13,14,15);
}

__global__ __launch_bounds__(256)
void Attention_28613072126828_kernel(const float* __restrict__ Qg,
                                     const float* __restrict__ Kg,
                                     const float* __restrict__ Vg,
                                     float* __restrict__ Og) {
    extern __shared__ char smem[];
    _Float16* Qs  = (_Float16*)smem;                       // [32][1024] f16   64KB
    _Float16* Ks  = Qs  + BM * DHEAD;                      // [32][1024] f16   64KB
    _Float16* VTs = Ks  + BN * DHEAD;                      // [1024][32] f16   64KB (V transposed)
    float*    Sc  = (float*)(VTs + DHEAD * BN);            // [32][32]  f32     4KB
    _Float16* Ps  = (_Float16*)(Sc + BM * BN);             // [32][32]  f16     2KB
    float*    m_run    = (float*)(Ps + BM * BN);           // [32]
    float*    l_run    = m_run + BM;                       // [32]
    float*    rowscale = l_run + BM;                       // [32]

    const int tid  = threadIdx.x;
    const int lane = tid & 31;
    const int wave = tid >> 5;
    const int qb   = blockIdx.x;
    const int b    = blockIdx.y;
    const int q0   = qb * BM;

    const int hi8 = (lane >> 4) << 3;   // C-tile row offset for this lane half
    const int nlo = lane & 15;          // C-tile column for this lane

    // ---- load Q tile once (contiguous 32x1024 fp32 -> f16 LDS) ----
    {
        const float* Qp = Qg + ((size_t)b * SEQ + q0) * DHEAD;
        for (int i = tid; i < BM * DHEAD / 4; i += 256) {
            const float4 q4 = ((const float4*)Qp)[i];
            _Float16* dst = Qs + i * 4;
            dst[0] = (_Float16)q4.x; dst[1] = (_Float16)q4.y;
            dst[2] = (_Float16)q4.z; dst[3] = (_Float16)q4.w;
        }
    }
    for (int i = tid; i < BM * BN; i += 256) Sc[i] = 0.0f;
    if (tid < BM) { m_run[tid] = -1e30f; l_run[tid] = 0.0f; }

    // Output accumulators: this wave owns out columns [c0, c0+128)
    const int d0 = wave * 128;   // QK^T reduction slice
    const int c0 = wave * 128;   // PV output-column slice
    v8f Oacc[2][8];
    {
        v8f z = {};
        #pragma unroll
        for (int mt = 0; mt < 2; ++mt)
            #pragma unroll
            for (int nt = 0; nt < 8; ++nt) Oacc[mt][nt] = z;
    }

    for (int kb = 0; kb <= qb; ++kb) {
        const int k0 = kb * BN;
        const float* Kp = Kg + ((size_t)b * SEQ + k0) * DHEAD;
        const float* Vp = Vg + ((size_t)b * SEQ + k0) * DHEAD;

        // ---- stream K tile (row-major) and V tile (transposed) into LDS ----
        for (int i = tid; i < BN * DHEAD / 4; i += 256) {
            const float4 k4 = ((const float4*)Kp)[i];
            _Float16* dst = Ks + i * 4;
            dst[0] = (_Float16)k4.x; dst[1] = (_Float16)k4.y;
            dst[2] = (_Float16)k4.z; dst[3] = (_Float16)k4.w;
        }
        for (int i = tid; i < BN * DHEAD / 4; i += 256) {
            const float4 v4 = ((const float4*)Vp)[i];
            const int key = (i * 4) / DHEAD;
            const int d   = (i * 4) % DHEAD;
            VTs[(d + 0) * BN + key] = (_Float16)v4.x;
            VTs[(d + 1) * BN + key] = (_Float16)v4.y;
            VTs[(d + 2) * BN + key] = (_Float16)v4.z;
            VTs[(d + 3) * BN + key] = (_Float16)v4.w;
        }
        __syncthreads();

        // ---- QK^T partial over this wave's 128-wide D slice ----
        v8f s_acc[2][2];
        {
            v8f z = {};
            s_acc[0][0] = z; s_acc[0][1] = z; s_acc[1][0] = z; s_acc[1][1] = z;
        }
        #pragma unroll
        for (int kc = 0; kc < 4; ++kc) {
            const int d = d0 + kc * 32;
            v16h a0 = load_frag_a(Qs,  0, DHEAD, d, lane);
            v16h a1 = load_frag_a(Qs, 16, DHEAD, d, lane);
            v16h b0 = load_frag_b(Ks,  0, DHEAD, d, lane);
            v16h b1 = load_frag_b(Ks, 16, DHEAD, d, lane);
            s_acc[0][0] = wmma_f16(a0, b0, s_acc[0][0]);
            s_acc[0][1] = wmma_f16(a0, b1, s_acc[0][1]);
            s_acc[1][0] = wmma_f16(a1, b0, s_acc[1][0]);
            s_acc[1][1] = wmma_f16(a1, b1, s_acc[1][1]);
        }
        // merge the 8 wave-partials into the LDS score tile (ds_add_f32)
        #pragma unroll
        for (int mt = 0; mt < 2; ++mt)
            #pragma unroll
            for (int nt = 0; nt < 2; ++nt)
                #pragma unroll
                for (int v = 0; v < 8; ++v)
                    atomicAdd(&Sc[(mt * 16 + hi8 + v) * BN + nt * 16 + nlo],
                              s_acc[mt][nt][v]);
        __syncthreads();

        // ---- online softmax: wave w owns rows 4w..4w+3, lanes own columns ----
        {
            const int c = lane;
            #pragma unroll
            for (int rr = 0; rr < 4; ++rr) {
                const int r = wave * 4 + rr;
                const int qrow = q0 + r;
                float x = Sc[r * BN + c] * 0.03125f;   // 1/sqrt(1024)
                if (k0 + c > qrow) x = -1e30f;         // causal mask
                // wave32 butterfly max-reduction over the 32 columns
                float mblk = x;
                #pragma unroll
                for (int off = 16; off > 0; off >>= 1)
                    mblk = fmaxf(mblk, __shfl_xor(mblk, off, 32));
                const float mold = m_run[r];
                const float mnew = fmaxf(mold, mblk);
                const float p = __expf(x - mnew);
                Ps[r * BN + c] = (_Float16)p;
                float lsum = p;
                #pragma unroll
                for (int off = 16; off > 0; off >>= 1)
                    lsum += __shfl_xor(lsum, off, 32);
                if (lane == 0) {
                    const float corr = __expf(mold - mnew);
                    l_run[r]    = l_run[r] * corr + lsum;
                    m_run[r]    = mnew;
                    rowscale[r] = corr;
                }
            }
        }
        __syncthreads();

        // re-zero score tile for the next iteration
        for (int i = tid; i < BM * BN; i += 256) Sc[i] = 0.0f;

        // ---- rescale running O by the softmax correction ----
        #pragma unroll
        for (int mt = 0; mt < 2; ++mt) {
            #pragma unroll
            for (int v = 0; v < 8; ++v) {
                const float f = rowscale[mt * 16 + hi8 + v];
                #pragma unroll
                for (int nt = 0; nt < 8; ++nt) Oacc[mt][nt][v] *= f;
            }
        }

        // ---- P x V into this wave's 128 output columns ----
        v16h pa0 = load_frag_a(Ps,  0, BN, 0, lane);
        v16h pa1 = load_frag_a(Ps, 16, BN, 0, lane);
        #pragma unroll
        for (int nt = 0; nt < 8; ++nt) {
            v16h vb = load_frag_b(VTs, c0 + nt * 16, BN, 0, lane);
            Oacc[0][nt] = wmma_f16(pa0, vb, Oacc[0][nt]);
            Oacc[1][nt] = wmma_f16(pa1, vb, Oacc[1][nt]);
        }

        // prefetch next K/V tiles into L2 (global_prefetch_b8)
        if (kb < qb) {
            const size_t nxt = ((size_t)b * SEQ + k0 + BN) * DHEAD;
            __builtin_prefetch(Kg + nxt + (size_t)tid * 128, 0, 0);
            __builtin_prefetch(Vg + nxt + (size_t)tid * 128, 0, 0);
        }
        __syncthreads();
    }

    // ---- epilogue: divide by softmax denominator, store fp32 ----
    #pragma unroll
    for (int mt = 0; mt < 2; ++mt) {
        #pragma unroll
        for (int v = 0; v < 8; ++v) {
            const int row = mt * 16 + hi8 + v;
            const float invl = 1.0f / l_run[row];
            float* op = Og + ((size_t)b * SEQ + q0 + row) * DHEAD + c0 + nlo;
            #pragma unroll
            for (int nt = 0; nt < 8; ++nt)
                op[nt * 16] = Oacc[mt][nt][v] * invl;
        }
    }
}

extern "C" void kernel_launch(void* const* d_in, const int* in_sizes, int n_in,
                              void* d_out, int out_size, void* d_ws, size_t ws_size,
                              hipStream_t stream) {
    (void)in_sizes; (void)n_in; (void)out_size; (void)d_ws; (void)ws_size;
    const float* Q = (const float*)d_in[0];
    const float* K = (const float*)d_in[1];
    const float* V = (const float*)d_in[2];
    float* O = (float*)d_out;

    const size_t lds_bytes =
        (size_t)(3 * BM * DHEAD) * sizeof(_Float16)   // Q, K, V^T tiles
        + (size_t)(BM * BN) * sizeof(float)           // score tile
        + (size_t)(BM * BN) * sizeof(_Float16)        // prob tile
        + (size_t)(3 * BM) * sizeof(float);           // m, l, rowscale

    // Idempotent, host-side, not stream-ordered: safe under graph capture,
    // no call-count-dependent behavior.
    hipFuncSetAttribute((const void*)Attention_28613072126828_kernel,
                        hipFuncAttributeMaxDynamicSharedMemorySize,
                        (int)lds_bytes);

    dim3 grid(SEQ / BM, 8);   // (query blocks, batch)
    dim3 block(256);          // 8 wave32s
    Attention_28613072126828_kernel<<<grid, block, lds_bytes, stream>>>(Q, K, V, O);
}
